// AttentionInterRNN_32787780337868
// MI455X (gfx1250) — compile-verified, matrix operand order checked by reading, not run on hardware
//
#include <hip/hip_runtime.h>
#include <hip/hip_bf16.h>
#include <stdint.h>

// Problem sizes (fixed by the reference)
#define TT 256
#define BB 64
#define II 1024
#define HH 512
#define CC 32
#define G3 (3 * HH)      // 1536
#define O2 (2 * HH)      // 1024
#define TB (TT * BB)     // 16384

#define BPAD 40          // padded LDS row length (elements) -> 80B, conflict-free frag reads

typedef __bf16 bf16_t;
typedef __attribute__((ext_vector_type(16))) __bf16 v16bf;
typedef __attribute__((ext_vector_type(8)))  float  v8f;

// ---------- helpers ----------

__device__ __forceinline__ bf16_t f2bf(float f) {
    union { float f; unsigned u; } v; v.f = f;
    unsigned r = v.u + 0x7FFFu + ((v.u >> 16) & 1u);   // round-to-nearest-even
    unsigned short h = (unsigned short)(r >> 16);
    return __builtin_bit_cast(bf16_t, h);
}

// Async 16B global -> LDS copy (gfx1250, ASYNCcnt-tracked)
__device__ __forceinline__ void async_copy16(unsigned lds_off, const void* gptr) {
    asm volatile("global_load_async_to_lds_b128 %0, %1, off"
                 :: "v"(lds_off), "v"(gptr) : "memory");
}
__device__ __forceinline__ void wait_async0() {
    asm volatile("s_wait_asynccnt 0x0" ::: "memory");
}

// Load one 16x32 bf16 A/B fragment row-slice from GLOBAL for this lane.
// p points at: row_base + k0 + khalf*8 (two contiguous 8-elem chunks, 16 apart)
__device__ __forceinline__ v16bf load_frag(const bf16_t* __restrict__ p) {
    v16bf f;
#pragma unroll
    for (int i = 0; i < 8; ++i) { f[i] = p[i]; f[i + 8] = p[i + 16]; }
    return f;
}

// Load one fragment row-slice from an LDS-staged B tile (row-major, BPAD stride)
__device__ __forceinline__ v16bf lds_frag(const bf16_t* sm, int row, int khalf) {
    const bf16_t* p = sm + row * BPAD + khalf * 8;
    v16bf f;
#pragma unroll
    for (int i = 0; i < 8; ++i) { f[i] = p[i]; f[i + 8] = p[i + 16]; }
    return f;
}

// ---------- conversion / init kernels ----------

__global__ void k_cvt(const float* __restrict__ src, bf16_t* __restrict__ dst, int n) {
    int i = blockIdx.x * blockDim.x + threadIdx.x;
    if (i < n) dst[i] = f2bf(src[i]);
}

// W_inter is [2H(in), 2H(out)] ; we want Wt[out][in] row-major (N,K form)
__global__ void k_cvt_transpose(const float* __restrict__ W, bf16_t* __restrict__ Wt) {
    int i = blockIdx.x * blockDim.x + threadIdx.x;
    if (i >= O2 * O2) return;
    int o = i / O2, in = i % O2;
    Wt[(size_t)o * O2 + in] = f2bf(W[(size_t)in * O2 + o]);
}

__global__ void k_init_h(const float* __restrict__ state, float* __restrict__ hf,
                         bf16_t* __restrict__ hb) {
    int i = blockIdx.x * blockDim.x + threadIdx.x;
    if (i >= 2 * BB * HH) return;
    float v = state[i];
    hf[i] = v;              // parity-0 region
    hb[i] = f2bf(v);
}

__global__ void k_zero(float* __restrict__ p, int n) {
    int i = blockIdx.x * blockDim.x + threadIdx.x;
    if (i < n) p[i] = 0.f;
}

// ---------- big GEMM: xg = x @ w_ih^T + b_ih  (M=TB, N=3H, K=I) ----------
// block = 4 waves; wave computes 16 rows x 64 cols (4 accum tiles).
// B tile (64 x 32, shared by all 4 waves) is staged in LDS via async copies.

__global__ __launch_bounds__(128) void k_gemm_xg(const bf16_t* __restrict__ xbf,
                                                 const bf16_t* __restrict__ w,
                                                 const float* __restrict__ bias,
                                                 float* __restrict__ xg, int reverse) {
    __shared__ bf16_t smB[64 * BPAD];
    const unsigned smB_base = (unsigned)(uintptr_t)&smB[0];

    const int tid   = threadIdx.x;
    const int lane  = tid & 31;
    const int wv    = tid >> 5;
    const int lr    = lane & 15;
    const int khalf = lane >> 4;
    const int m0    = (blockIdx.x * 4 + wv) * 16;
    const int n0    = blockIdx.y * 64;

    int mrow = m0 + lr;
    if (reverse) { int t = mrow / BB, b = mrow % BB; mrow = (TT - 1 - t) * BB + b; }
    const bf16_t* aptr = xbf + (size_t)mrow * II + khalf * 8;

    v8f acc[4] = {v8f{}, v8f{}, v8f{}, v8f{}};

    for (int k0 = 0; k0 < II; k0 += 32) {
        // stage 64x32 B tile: 256 16B chunks, 2 per thread
#pragma unroll
        for (int c = 0; c < 2; ++c) {
            int chunk = tid * 2 + c;
            int row = chunk >> 2, sub = chunk & 3;
            const bf16_t* g = w + (size_t)(n0 + row) * II + k0 + sub * 8;
            async_copy16(smB_base + (unsigned)(row * BPAD + sub * 8) * 2u, g);
        }
        wait_async0();
        __syncthreads();

        v16bf afrag = load_frag(aptr + k0);
#pragma unroll
        for (int nt = 0; nt < 4; ++nt) {
            v16bf bfrag = lds_frag(smB, nt * 16 + lr, khalf);
            acc[nt] = __builtin_amdgcn_wmma_f32_16x16x32_bf16(
                false, afrag, false, bfrag, (short)0, acc[nt], false, false);
        }
        __syncthreads();
    }
#pragma unroll
    for (int nt = 0; nt < 4; ++nt) {
        int n = n0 + nt * 16 + lr;
        float bv = bias[n];
#pragma unroll
        for (int r = 0; r < 8; ++r) {
            int row = m0 + khalf * 8 + r;               // logical (t*B+b) row
            xg[(size_t)row * G3 + n] = acc[nt][r] + bv;
        }
    }
}

// ---------- GRU step: hg = h @ w_hh^T + b_hh ; gate math ; write y ----------
// grid: (H/64 j-tiles, 2 dirs), block = 4 waves (each wave = 16 of the 64 batch rows)
// h double-buffered by time parity: layout [parity][dir][B][H]
// w_hh tile (3 gates x 64 cols x 32 k, shared by all waves) staged in LDS async.

__global__ __launch_bounds__(128) void k_gru_step(
    const bf16_t* __restrict__ whh_f, const bf16_t* __restrict__ whh_b,
    const float* __restrict__ bhh_f,  const float* __restrict__ bhh_b,
    const float* __restrict__ xg_f,   const float* __restrict__ xg_b,
    float* __restrict__ h_f32, bf16_t* __restrict__ h_bf16,
    float* __restrict__ out_f32, bf16_t* __restrict__ out_bf16, int t) {

    __shared__ bf16_t smB[192 * BPAD];
    const unsigned smB_base = (unsigned)(uintptr_t)&smB[0];

    const int dir = blockIdx.y;
    const int j0  = blockIdx.x * 64;
    const bf16_t* whh = dir ? whh_b : whh_f;
    const float*  bhh = dir ? bhh_b : bhh_f;
    const float*  xg  = (dir ? xg_b : xg_f) + (size_t)t * BB * G3;

    const size_t pstride = (size_t)2 * BB * HH;
    const float*  hrd_f = h_f32  + (size_t)(t & 1) * pstride        + (size_t)dir * BB * HH;
    const bf16_t* hrd_b = h_bf16 + (size_t)(t & 1) * pstride        + (size_t)dir * BB * HH;
    float*        hwr_f = h_f32  + (size_t)((t + 1) & 1) * pstride  + (size_t)dir * BB * HH;
    bf16_t*       hwr_b = h_bf16 + (size_t)((t + 1) & 1) * pstride  + (size_t)dir * BB * HH;

    const int tid   = threadIdx.x;
    const int lane  = tid & 31;
    const int wv    = tid >> 5;
    const int lr    = lane & 15;
    const int khalf = lane >> 4;
    const int m0    = wv * 16;                 // batch-row tile

    v8f acc[3][4];
#pragma unroll
    for (int g = 0; g < 3; ++g)
#pragma unroll
        for (int nt = 0; nt < 4; ++nt) acc[g][nt] = v8f{};

    const bf16_t* aptr = hrd_b + (size_t)(m0 + lr) * HH + khalf * 8;

    for (int k0 = 0; k0 < HH; k0 += 32) {
        // stage 192x32 B tile: 768 16B chunks, 6 per thread
#pragma unroll
        for (int c = 0; c < 6; ++c) {
            int chunk = tid * 6 + c;
            int row = chunk >> 2, sub = chunk & 3;
            int grow = (row >> 6) * HH + j0 + (row & 63);
            const bf16_t* g = whh + (size_t)grow * HH + k0 + sub * 8;
            async_copy16(smB_base + (unsigned)(row * BPAD + sub * 8) * 2u, g);
        }
        wait_async0();
        __syncthreads();

        v16bf afrag = load_frag(aptr + k0);
#pragma unroll
        for (int g = 0; g < 3; ++g) {
#pragma unroll
            for (int nt = 0; nt < 4; ++nt) {
                v16bf bfrag = lds_frag(smB, g * 64 + nt * 16 + lr, khalf);
                acc[g][nt] = __builtin_amdgcn_wmma_f32_16x16x32_bf16(
                    false, afrag, false, bfrag, (short)0, acc[g][nt], false, false);
            }
        }
        __syncthreads();
    }

    const int tout = dir ? (TT - 1 - t) : t;
#pragma unroll
    for (int nt = 0; nt < 4; ++nt) {
        int j = j0 + nt * 16 + lr;
        float br = bhh[j], bz = bhh[HH + j], bn = bhh[2 * HH + j];
#pragma unroll
        for (int r = 0; r < 8; ++r) {
            int b = m0 + khalf * 8 + r;
            const float* xrow = xg + (size_t)b * G3;
            float hr = acc[0][nt][r] + br;
            float hz = acc[1][nt][r] + bz;
            float hn = acc[2][nt][r] + bn;
            float rg = 1.f / (1.f + __expf(-(xrow[j] + hr)));
            float zg = 1.f / (1.f + __expf(-(xrow[HH + j] + hz)));
            float ng = tanhf(xrow[2 * HH + j] + rg * hn);
            float hold = hrd_f[(size_t)b * HH + j];
            float h2 = (1.f - zg) * ng + zg * hold;
            hwr_f[(size_t)b * HH + j] = h2;
            hwr_b[(size_t)b * HH + j] = f2bf(h2);
            size_t o = ((size_t)tout * BB + b) * O2 + (dir ? HH + j : j);
            out_f32[o] = h2;
            out_bf16[o] = f2bf(h2);
        }
    }
}

// ---------- fused squish GEMM + tanh + weight_proj reduction -> attn ----------
// M=TB rows of out_bf16, N=K=2H. attn[row] += sum_n tanh(acc+bias)*wp[n] via atomics.
// B tile (W_inter^T 64x32) staged in LDS via async copies.

__global__ __launch_bounds__(128) void k_attn(const bf16_t* __restrict__ outbf,
                                              const bf16_t* __restrict__ WinterT,
                                              const float* __restrict__ bias_inter,
                                              const float* __restrict__ wproj,
                                              float* __restrict__ attn) {
    __shared__ bf16_t smB[64 * BPAD];
    const unsigned smB_base = (unsigned)(uintptr_t)&smB[0];

    const int tid   = threadIdx.x;
    const int lane  = tid & 31;
    const int wv    = tid >> 5;
    const int lr    = lane & 15;
    const int khalf = lane >> 4;
    const int m0    = (blockIdx.x * 4 + wv) * 16;
    const int n0    = blockIdx.y * 64;

    v8f acc[4] = {v8f{}, v8f{}, v8f{}, v8f{}};
    const bf16_t* aptr = outbf + (size_t)(m0 + lr) * O2 + khalf * 8;

    for (int k0 = 0; k0 < O2; k0 += 32) {
#pragma unroll
        for (int c = 0; c < 2; ++c) {
            int chunk = tid * 2 + c;
            int row = chunk >> 2, sub = chunk & 3;
            const bf16_t* g = WinterT + (size_t)(n0 + row) * O2 + k0 + sub * 8;
            async_copy16(smB_base + (unsigned)(row * BPAD + sub * 8) * 2u, g);
        }
        wait_async0();
        __syncthreads();

        v16bf afrag = load_frag(aptr + k0);
#pragma unroll
        for (int nt = 0; nt < 4; ++nt) {
            v16bf bfrag = lds_frag(smB, nt * 16 + lr, khalf);
            acc[nt] = __builtin_amdgcn_wmma_f32_16x16x32_bf16(
                false, afrag, false, bfrag, (short)0, acc[nt], false, false);
        }
        __syncthreads();
    }

    float rs[8] = {0.f, 0.f, 0.f, 0.f, 0.f, 0.f, 0.f, 0.f};
#pragma unroll
    for (int nt = 0; nt < 4; ++nt) {
        int n = n0 + nt * 16 + lr;
        float bi = bias_inter[n];
        float wp = wproj[n];
#pragma unroll
        for (int r = 0; r < 8; ++r) rs[r] += tanhf(acc[nt][r] + bi) * wp;
    }
    // reduce across the 16 lanes that share each output row (xor masks stay in half)
#pragma unroll
    for (int m = 8; m >= 1; m >>= 1)
#pragma unroll
        for (int r = 0; r < 8; ++r) rs[r] += __shfl_xor(rs[r], m, 32);
    if (lr == 0) {
#pragma unroll
        for (int r = 0; r < 8; ++r)
            atomicAdd(&attn[m0 + khalf * 8 + r], rs[r]);
    }
}

// ---------- pooled / final / outputs ----------

__global__ void k_pool(const float* __restrict__ attn, const float* __restrict__ outf,
                       float* __restrict__ pooled) {
    int idx = blockIdx.x * blockDim.x + threadIdx.x;
    if (idx >= BB * O2) return;
    int b = idx / O2, hcol = idx % O2;
    float s = 0.f;
    for (int t = 0; t < TT; ++t)
        s += attn[t * BB + b] * outf[((size_t)t * BB + b) * O2 + hcol];
    pooled[idx] = s;
}

__global__ void k_final(const float* __restrict__ pooled, const float* __restrict__ Wf,
                        const float* __restrict__ bf, float* __restrict__ out) {
    int idx = blockIdx.x * blockDim.x + threadIdx.x;
    if (idx >= BB * CC) return;
    int b = idx / CC, c = idx % CC;
    float s = bf[c];
    const float* pr = pooled + (size_t)b * O2;
    const float* wr = Wf + (size_t)c * O2;
    for (int k = 0; k < O2; ++k) s += pr[k] * wr[k];
    out[idx] = s;       // final_map at offset 0
}

__global__ void k_state_out(const float* __restrict__ hstate, float* __restrict__ out) {
    int i = blockIdx.x * blockDim.x + threadIdx.x;
    if (i < 2 * BB * HH) out[BB * CC + i] = hstate[i];     // parity-0 region holds final h
}

__global__ void k_attn_out(const float* __restrict__ attn, float* __restrict__ out) {
    int i = blockIdx.x * blockDim.x + threadIdx.x;
    if (i >= BB * TT) return;
    int b = i / TT, t = i % TT;
    out[BB * CC + 2 * BB * HH + i] = attn[t * BB + b];     // attn.T [B,T]
}

// ---------- launcher ----------

extern "C" void kernel_launch(void* const* d_in, const int* in_sizes, int n_in,
                              void* d_out, int out_size, void* d_ws, size_t ws_size,
                              hipStream_t stream) {
    const float* x        = (const float*)d_in[0];
    const float* state    = (const float*)d_in[1];
    const float* w_ih_f   = (const float*)d_in[2];
    const float* w_hh_f   = (const float*)d_in[3];
    const float* b_ih_f   = (const float*)d_in[4];
    const float* b_hh_f   = (const float*)d_in[5];
    const float* w_ih_b   = (const float*)d_in[6];
    const float* w_hh_b   = (const float*)d_in[7];
    const float* b_ih_b   = (const float*)d_in[8];
    const float* b_hh_b   = (const float*)d_in[9];
    const float* W_inter  = (const float*)d_in[10];
    const float* bias_int = (const float*)d_in[11];
    const float* wproj    = (const float*)d_in[12];
    const float* W_final  = (const float*)d_in[13];
    const float* b_final  = (const float*)d_in[14];
    float* out = (float*)d_out;

    // workspace carve-up (256B aligned)
    char* ws = (char*)d_ws;
    size_t off = 0;
    auto carve = [&](size_t bytes) -> char* {
        char* p = ws + off;
        off = (off + bytes + 255) & ~(size_t)255;
        return p;
    };
    bf16_t* xbf      = (bf16_t*)carve((size_t)TB * II * 2);
    bf16_t* wihf_bf  = (bf16_t*)carve((size_t)G3 * II * 2);
    bf16_t* wihb_bf  = (bf16_t*)carve((size_t)G3 * II * 2);
    bf16_t* whhf_bf  = (bf16_t*)carve((size_t)G3 * HH * 2);
    bf16_t* whhb_bf  = (bf16_t*)carve((size_t)G3 * HH * 2);
    bf16_t* wintT_bf = (bf16_t*)carve((size_t)O2 * O2 * 2);
    float*  xg_f     = (float*)carve((size_t)TB * G3 * 4);
    float*  xg_b     = (float*)carve((size_t)TB * G3 * 4);
    float*  out_f32  = (float*)carve((size_t)TB * O2 * 4);
    bf16_t* out_bf16 = (bf16_t*)carve((size_t)TB * O2 * 2);
    float*  h_f32    = (float*)carve((size_t)2 * 2 * BB * HH * 4);
    bf16_t* h_bf16   = (bf16_t*)carve((size_t)2 * 2 * BB * HH * 2);
    float*  attn     = (float*)carve((size_t)TB * 4);
    float*  pooled   = (float*)carve((size_t)BB * O2 * 4);
    (void)ws_size; (void)in_sizes; (void)n_in; (void)out_size;

    const int TPB = 256;
    auto blocks = [&](long n) { return dim3((unsigned)((n + TPB - 1) / TPB)); };

    // 1. conversions + init
    k_cvt<<<blocks((long)TB * II), TPB, 0, stream>>>(x, xbf, TB * II);
    k_cvt<<<blocks((long)G3 * II), TPB, 0, stream>>>(w_ih_f, wihf_bf, G3 * II);
    k_cvt<<<blocks((long)G3 * II), TPB, 0, stream>>>(w_ih_b, wihb_bf, G3 * II);
    k_cvt<<<blocks((long)G3 * HH), TPB, 0, stream>>>(w_hh_f, whhf_bf, G3 * HH);
    k_cvt<<<blocks((long)G3 * HH), TPB, 0, stream>>>(w_hh_b, whhb_bf, G3 * HH);
    k_cvt_transpose<<<blocks((long)O2 * O2), TPB, 0, stream>>>(W_inter, wintT_bf);
    k_init_h<<<blocks(2 * BB * HH), TPB, 0, stream>>>(state, h_f32, h_bf16);
    k_zero<<<blocks(TB), TPB, 0, stream>>>(attn, TB);

    // 2. input-gate GEMMs (both directions)
    dim3 gx(TB / 64, G3 / 64);
    k_gemm_xg<<<gx, 128, 0, stream>>>(xbf, wihf_bf, b_ih_f, xg_f, 0);
    k_gemm_xg<<<gx, 128, 0, stream>>>(xbf, wihb_bf, b_ih_b, xg_b, 1);

    // 3. recurrent scan: one launch per timestep (launch boundary = global sync)
    dim3 gs(HH / 64, 2);
    for (int t = 0; t < TT; ++t) {
        k_gru_step<<<gs, 128, 0, stream>>>(whhf_bf, whhb_bf, b_hh_f, b_hh_b,
                                           xg_f, xg_b, h_f32, h_bf16,
                                           out_f32, out_bf16, t);
    }

    // 4. fused attention squish + projection
    dim3 ga(TB / 64, O2 / 64);
    k_attn<<<ga, 128, 0, stream>>>(out_bf16, wintT_bf, bias_int, wproj, attn);

    // 5. pooling + outputs
    k_pool<<<blocks((long)BB * O2), TPB, 0, stream>>>(attn, out_f32, pooled);
    k_final<<<blocks(BB * CC), TPB, 0, stream>>>(pooled, W_final, b_final, out);
    k_state_out<<<blocks(2 * BB * HH), TPB, 0, stream>>>(h_f32, out);
    k_attn_out<<<blocks(BB * TT), TPB, 0, stream>>>(attn, out);
}
